// SoftNNLossPos_44160853738083
// MI455X (gfx1250) — compile-verified, matrix operand order checked by reading, not run on hardware
//
#include <hip/hip_runtime.h>
#include <hip/hip_bf16.h>

typedef __attribute__((ext_vector_type(16))) __bf16 v16bf;
typedef __attribute__((ext_vector_type(8)))  float  v8f;

#define kN 4096
#define kM 4096
#define kK 32768
#define kD 1024
#define kRowTile 256            /* block rows */
#define kColTile 128            /* block cols */
#define kKStep 32
#define kLdsStride 40           /* 32 bf16 + 8 pad -> 80B pitch, conflict-free b128 */
#define kAElems (kRowTile * kLdsStride)   /* 10240 elems per A buffer */
#define kBElems (kColTile * kLdsStride)   /*  5120 elems per B buffer */
#define kSmemElems (3 * kAElems + 3 * kBElems)         /* triple buffered */
#define kSmemBytes (kSmemElems * 2)                    /* 92160 B = 90 KB  */

// Workspace layout (bytes)
#define WS_IMG  ((size_t)0)
#define WS_POS  ((size_t)kN * kD * 2)                 // 8 MiB
#define WS_OTH  (WS_POS + (size_t)kM * kD * 2)        // 16 MiB
#define WS_PSUM (WS_OTH + (size_t)kK * kD * 2)        // 80 MiB
#define WS_DEN  (WS_PSUM + (size_t)kN * 4)

static __device__ __forceinline__ unsigned short f32_to_bf16_rne(float f) {
    unsigned int u = __float_as_uint(f);
    u += 0x7FFFu + ((u >> 16) & 1u);   // round to nearest even
    return (unsigned short)(u >> 16);
}

// ---------------- fp32 -> bf16 conversion (vectorized x4) ----------------
__global__ void snnl_convert_bf16(const float* __restrict__ src,
                                  unsigned short* __restrict__ dst, int n4) {
    int i = blockIdx.x * blockDim.x + threadIdx.x;
    if (i >= n4) return;
    float4 f = ((const float4*)src)[i];
    ushort4 o;
    o.x = f32_to_bf16_rne(f.x);
    o.y = f32_to_bf16_rne(f.y);
    o.z = f32_to_bf16_rne(f.z);
    o.w = f32_to_bf16_rne(f.w);
    ((ushort4*)dst)[i] = o;
}

// ---------------- zero-init for the row-sum accumulators ----------------
__global__ void snnl_zero(float* __restrict__ p, int n) {
    int i = blockIdx.x * blockDim.x + threadIdx.x;
    if (i < n) p[i] = 0.0f;
}

// ---------------- fused bf16-WMMA GEMM + exp + row-sum ----------------
// rowsums[r] += sum_c exp(scale * dot(A[r,:], B[c,:]))
// Grid: (cols/128, rows/256). Block: 256 threads = 8 wave32
// (4 row-waves x 2 col-waves); each wave computes a 64x64 tile
// (4 row-frags x 4 col-frags = 16 WMMAs per k-step).
// Global->LDS staging uses CDNA5 async copies (ASYNCcnt), triple-buffered
// with distance-1 prefetch so a single barrier per k-step is race-free.
__global__ __launch_bounds__(256)
void snnl_gemm_exp_rowsum(const unsigned short* __restrict__ Abf,
                          const unsigned short* __restrict__ Bbf,
                          float* __restrict__ rowsums, float scale) {
    extern __shared__ uint4 smem4[];                       // 16B-aligned dynamic LDS
    unsigned short* smem = (unsigned short*)smem4;
    const unsigned lds_base = (unsigned)(size_t)smem;      // low 32 bits = LDS offset

    const int tid  = threadIdx.x;
    const int lane = tid & 31;
    const int wave = tid >> 5;
    const int wr   = wave & 3;   // row-wave: 64 rows each
    const int wc   = wave >> 2;  // col-wave: 64 cols each
    const int hi   = lane >> 4;  // lane half (ISA bf16 fragment layout)
    const int lrow = lane & 15;

    const long rowbase = (long)blockIdx.y * kRowTile;
    const long colbase = (long)blockIdx.x * kColTile;

    v8f acc[4][4];
#pragma unroll
    for (int rf = 0; rf < 4; ++rf)
#pragma unroll
        for (int cf = 0; cf < 4; ++cf)
#pragma unroll
            for (int e = 0; e < 8; ++e) acc[rf][cf][e] = 0.0f;

    // Async-stage a 256x32 (A) + 128x32 (B) bf16 tile into LDS buffer `buf`.
    // 6 async b128 copies per thread (A: 4 chunks, B: 2 chunks).
    auto stage = [&](int kt, int buf) {
        const long k0 = (long)kt * kKStep;
        const unsigned baseA = lds_base + (unsigned)(buf * kAElems) * 2u;
        const unsigned baseB = lds_base + (unsigned)(3 * kAElems + buf * kBElems) * 2u;
#pragma unroll
        for (int i = 0; i < 4; ++i) {                    // A: 1024 chunks of 8 elems
            int c = tid + i * 256;
            int trow = c >> 2, koff = (c & 3) * 8;
            unsigned lo = baseA + (unsigned)(trow * kLdsStride + koff) * 2u;
            const unsigned short* g = Abf + (rowbase + trow) * (long)kD + k0 + koff;
            asm volatile("global_load_async_to_lds_b128 %0, %1, off"
                         :: "v"(lo), "v"(g) : "memory");
        }
#pragma unroll
        for (int i = 0; i < 2; ++i) {                    // B: 512 chunks of 8 elems
            int c = tid + i * 256;
            int trow = c >> 2, koff = (c & 3) * 8;
            unsigned lo = baseB + (unsigned)(trow * kLdsStride + koff) * 2u;
            const unsigned short* g = Bbf + (colbase + trow) * (long)kD + k0 + koff;
            asm volatile("global_load_async_to_lds_b128 %0, %1, off"
                         :: "v"(lo), "v"(g) : "memory");
        }
    };

    union FragU { uint4 u[2]; v16bf v; };

    const int KT = kD / kKStep;  // 32 k-steps
    stage(0, 0);
    for (int kt = 0; kt < KT; ++kt) {
        const int cur = kt % 3;
        if (kt + 1 < KT) {
            stage(kt + 1, (kt + 1) % 3);
            // newest 6 copies may be outstanding; everything older (incl. stage kt) done
            asm volatile("s_wait_asynccnt 6" ::: "memory");
        } else {
            asm volatile("s_wait_asynccnt 0" ::: "memory");
        }
        __syncthreads();   // own async copies drained above => barrier implies tile ready

        const unsigned short* tileA = smem + cur * kAElems;
        const unsigned short* tileB = smem + 3 * kAElems + cur * kBElems;

        // Fragment assembly per ISA 16-bit layout: lane(l<16) K=c..c+7,c+16..c+23 (c=8*hi)
        FragU a[4], b[4];
#pragma unroll
        for (int rf = 0; rf < 4; ++rf) {
            int r = wr * 64 + rf * 16 + lrow;
            const unsigned short* p = tileA + r * kLdsStride + hi * 8;
            a[rf].u[0] = *(const uint4*)(p);
            a[rf].u[1] = *(const uint4*)(p + 16);
        }
#pragma unroll
        for (int cf = 0; cf < 4; ++cf) {
            int r = wc * 64 + cf * 16 + lrow;
            const unsigned short* p = tileB + r * kLdsStride + hi * 8;
            b[cf].u[0] = *(const uint4*)(p);
            b[cf].u[1] = *(const uint4*)(p + 16);
        }
#pragma unroll
        for (int rf = 0; rf < 4; ++rf)
#pragma unroll
            for (int cf = 0; cf < 4; ++cf)
                acc[rf][cf] = __builtin_amdgcn_wmma_f32_16x16x32_bf16(
                    false, a[rf].v, false, b[cf].v, (short)0, acc[rf][cf],
                    false, false);
    }

    // Epilogue: exp(scale*x), reduce across columns (16-lane halves), atomic row-sum.
#pragma unroll
    for (int rf = 0; rf < 4; ++rf) {
        v8f s;
#pragma unroll
        for (int e = 0; e < 8; ++e) s[e] = 0.0f;
#pragma unroll
        for (int cf = 0; cf < 4; ++cf)
#pragma unroll
            for (int e = 0; e < 8; ++e)
                s[e] += __expf(scale * acc[rf][cf][e]);
#pragma unroll
        for (int e = 0; e < 8; ++e) {
            float v = s[e];
            v += __shfl_xor(v, 1);
            v += __shfl_xor(v, 2);
            v += __shfl_xor(v, 4);
            v += __shfl_xor(v, 8);   // stays within each 16-lane half
            s[e] = v;
        }
        if (lrow == 0) {  // lanes 0 and 16 commit their half's 8 rows
#pragma unroll
            for (int e = 0; e < 8; ++e) {
                long row = rowbase + wr * 64 + rf * 16 + hi * 8 + e;
                atomicAdd(&rowsums[row], s[e]);
            }
        }
    }
}

// ---------------- final scalar loss reduction ----------------
// loss = -sum(log(pos_sum/denom)) = sum(log(denom) - log(pos_sum))
__global__ void snnl_loss(const float* __restrict__ pos_sum,
                          const float* __restrict__ denom,
                          float* __restrict__ out) {
    __shared__ float red[256];
    int tid = threadIdx.x;
    float s = 0.0f;
    for (int r = tid; r < kN; r += 256)
        s += __logf(denom[r]) - __logf(pos_sum[r]);
    red[tid] = s;
    __syncthreads();
    for (int off = 128; off > 0; off >>= 1) {
        if (tid < off) red[tid] += red[tid + off];
        __syncthreads();
    }
    if (tid == 0) out[0] = red[0];
}

extern "C" void kernel_launch(void* const* d_in, const int* in_sizes, int n_in,
                              void* d_out, int out_size, void* d_ws, size_t ws_size,
                              hipStream_t stream) {
    const float* img = (const float*)d_in[0];  // [N, D]
    const float* pos = (const float*)d_in[1];  // [M, D]
    const float* oth = (const float*)d_in[2];  // [K, D]

    char* ws = (char*)d_ws;
    unsigned short* imgb = (unsigned short*)(ws + WS_IMG);
    unsigned short* posb = (unsigned short*)(ws + WS_POS);
    unsigned short* othb = (unsigned short*)(ws + WS_OTH);
    float* psum = (float*)(ws + WS_PSUM);
    float* den  = (float*)(ws + WS_DEN);
    float* out  = (float*)d_out;

    // 1) fp32 -> bf16 (memory-bound; ~80 MB written)
    snnl_convert_bf16<<<(kN * kD / 4 + 255) / 256, 256, 0, stream>>>(img, imgb, kN * kD / 4);
    snnl_convert_bf16<<<(kM * kD / 4 + 255) / 256, 256, 0, stream>>>(pos, posb, kM * kD / 4);
    snnl_convert_bf16<<<(kK * kD / 4 + 255) / 256, 256, 0, stream>>>(oth, othb, kK * kD / 4);

    // 2) zero the row-sum accumulators (psum and den are contiguous: 2*N floats)
    snnl_zero<<<(2 * kN + 255) / 256, 256, 0, stream>>>(psum, 2 * kN);

    // 3) fused GEMM + exp + row-sum (compute-bound, bf16 WMMA + async LDS staging)
    snnl_gemm_exp_rowsum<<<dim3(kM / kColTile, kN / kRowTile), 256, kSmemBytes, stream>>>(
        imgb, posb, psum, 20.0f /* W/TEMPERATURE */);
    snnl_gemm_exp_rowsum<<<dim3(kK / kColTile, kN / kRowTile), 256, kSmemBytes, stream>>>(
        imgb, othb, den, 10.0f /* 1/TEMPERATURE */);

    // 4) scalar loss
    snnl_loss<<<1, 256, 0, stream>>>(psum, den, out);
}
